// VQVAE_50182397886712
// MI455X (gfx1250) — matrix operand order repaired and checked by reading, not compiled, not run on hardware
//
#include <hip/hip_runtime.h>
#include <hip/hip_bf16.h>
#include <math.h>

typedef __attribute__((ext_vector_type(16))) _Float16 v16h;
typedef __attribute__((ext_vector_type(8)))  _Float16 v8h;
typedef __attribute__((ext_vector_type(8)))  float    v8f;

#define WAVES 4   // 128-thread blocks = 4 wave32 waves

// CDNA5 16-bit A-fragment (16x32) element->K mapping:
// lanes 0-15 : element i -> K = (i<8 ? i : i+8)
// lanes 16-31: element i -> K = (i<8 ? i+8 : i+16)
// => element block [0,8)  covers K in [8*hi, 8*hi+8)        (contiguous)
//    element block [8,16) covers K in [16+8*hi, 16+8*hi+8)  (contiguous)
__device__ __forceinline__ int a_koff(int hi, int i) {
  return (i < 8 ? i : i + 8) + (hi << 3);
}

// ---------------------------------------------------------------------------
// FAST implicit-GEMM conv / convT for Cin % 32 == 0.
// Activations f16 NHWC; weights pre-repacked to fragment order (f16).
// Per wave: 16 pixels x NTL*16 out-channels. Always 4 independent
// accumulators: NTL output tiles x KS = 4/NTL K-split partials
// -> WMMA dependency distance >= 4, no RAW hazard NOPs for any variant.
// OOB taps are handled BRANCHLESSLY: the A-row pointer is cndmask-selected
// to a zeroed pad block, so all loads issue unconditionally and a whole
// chunk's A+B loads can be clustered in one s_clause with a single wait.
// act: 0 none, 1 relu, 2 sigmoid. outh/outf optional (f16 / f32 outputs).
// ---------------------------------------------------------------------------
template <int NTL>
__global__ void conv_fast_kernel(
    const _Float16* __restrict__ xh, const v16h* __restrict__ wp,
    const float* __restrict__ bias, const _Float16* __restrict__ zpad,
    _Float16* __restrict__ outh, float* __restrict__ outf,
    int Hin, int Win, int Cin, int Hout, int Wout, int Cout,
    int KH, int KW, int S, int pad, int transpose, int act)
{
  constexpr int KS = 4 / NTL;                 // K-split accumulators per tile

  const int lane = threadIdx.x & 31;
  const int wid  = threadIdx.x >> 5;
  const int hi   = lane >> 4;
  const int mr   = lane & 15;
  const int tileM = blockIdx.x * WAVES + wid;
  const int tn0   = blockIdx.y * NTL;         // first 16-wide N tile of this wave

  const int HWo = Hout * Wout;
  const int p   = tileM * 16 + mr;            // A-fragment row: output pixel
  const int bi  = p / HWo;
  const int rp  = p - bi * HWo;
  const int oh  = rp / Wout;
  const int ow  = rp - oh * Wout;

  const int Cin32  = Cin >> 5;
  const int chunks = KH * KW * Cin32;
  // per-j weight base pointers (lane-resident, stride 32 v16h per chunk)
  const v16h* wj[NTL];
#pragma unroll
  for (int j = 0; j < NTL; ++j)
    wj[j] = wp + ((size_t)(tn0 + j) * chunks) * 32 + lane;

  v8f acc[NTL][KS];
#pragma unroll
  for (int j = 0; j < NTL; ++j)
#pragma unroll
    for (int s = 0; s < KS; ++s) acc[j][s] = v8f{};

  int c = 0;
  for (int kh = 0; kh < KH; ++kh) {
    for (int kw = 0; kw < KW; ++kw) {
      // tap validity + input coords for this lane's pixel (shared by all ci blocks)
      int  ih, iw;
      bool ok;
      if (!transpose) {
        ih = oh * S + kh - pad;
        iw = ow * S + kw - pad;
        ok = (ih >= 0) && (ih < Hin) && (iw >= 0) && (iw < Win);
      } else {
        int th = oh + pad - kh;
        int tw = ow + pad - kw;
        ok = (th >= 0) && (tw >= 0) && (th % S == 0) && (tw % S == 0);
        ih = th / S;
        iw = tw / S;
        ok = ok && (ih >= 0) && (ih < Hin) && (iw >= 0) && (iw < Win);
      }
      // branchless: OOB lanes read from the zeroed pad block (>= Cin halves)
      const _Float16* base = xh + ((size_t)(bi * Hin + ih) * Win + iw) * Cin;
      const _Float16* arow = ok ? base : zpad;
      const int       cstp = ok ? 32 : 0;      // pad block is only 128 halves
      for (int cb = 0; cb < Cin32; ++cb, ++c) {
        // ---- load phase: A (2 x b128) + all B fragments (NTL x 32B) ----
        v8h lo = *(const v8h*)(arow + cb * cstp + 8 * hi);       // K [8hi, 8hi+8)
        v8h hh = *(const v8h*)(arow + cb * cstp + 16 + 8 * hi);  // K [16+8hi, +8)
        v16h a;
#pragma unroll
        for (int i = 0; i < 8; ++i) { a[i] = lo[i]; a[i + 8] = hh[i]; }
        v16h b[NTL];
#pragma unroll
        for (int j = 0; j < NTL; ++j) b[j] = wj[j][(size_t)c * 32];

        // ---- math phase: NTL independent WMMAs into K-split partials ----
        const int s = c & (KS - 1);
#pragma unroll
        for (int j = 0; j < NTL; ++j)
          acc[j][s] = __builtin_amdgcn_wmma_f32_16x16x32_f16(
              false, a, false, b[j], (short)0, acc[j][s], false, false);
      }
    }
  }

#pragma unroll
  for (int j = 0; j < NTL; ++j) {
    int   n  = (tn0 + j) * 16 + mr;
    float bb = (n < Cout) ? bias[n] : 0.f;
#pragma unroll
    for (int r = 0; r < 8; ++r) {
      int   m = tileM * 16 + r + (hi << 3);
      float v = bb;
#pragma unroll
      for (int s = 0; s < KS; ++s) v += acc[j][s][r];
      if (act == 1)      v = fmaxf(v, 0.f);
      else if (act == 2) v = 1.f / (1.f + expf(-v));
      if (n < Cout) {
        if (outh) outh[(size_t)m * Cout + n] = (_Float16)v;
        if (outf) outf[(size_t)m * Cout + n] = v;
      }
    }
  }
}

// ---------------------------------------------------------------------------
// Specialized conv1: 128x128x3 -> 64x64x32, 4x4 kernel, stride 2, pad 1, relu.
// KTOT = 48 -> exactly 2 chunks, fully unrolled; (kh,kw,ci) uses constant
// divisors; OOB elements load from the zero pad block through a per-element
// pointer cndmask, so all 24 gather loads issue unconditionally (clause-able).
// Two independent accumulators (one per chunk).
// ---------------------------------------------------------------------------
__global__ void conv1_kernel(const float* __restrict__ x, const v16h* __restrict__ wp,
                             const float* __restrict__ bias,
                             const float* __restrict__ zpadf,
                             _Float16* __restrict__ outh)
{
  constexpr int Hin = 128, Win = 128, Cin = 3, Cout = 32;
  constexpr int KW = 4, S = 2, PAD = 1, KTOT = 48, chunks = 2;

  const int lane = threadIdx.x & 31;
  const int wid  = threadIdx.x >> 5;
  const int hi   = lane >> 4;
  const int mr   = lane & 15;
  const int tileM = blockIdx.x * WAVES + wid;
  const int tileN = blockIdx.y;
  const int n     = tileN * 16 + mr;

  const int p  = tileM * 16 + mr;
  const int bi = p >> 12;               // / (64*64)
  const int rp = p & 4095;
  const int oh = rp >> 6;
  const int ow = rp & 63;

  const v16h* wrow = wp + (size_t)tileN * chunks * 32 + lane;
  v8f acc0 = {}, acc1 = {};

#pragma unroll
  for (int c = 0; c < chunks; ++c) {
    v16h a;
#pragma unroll
    for (int i = 0; i < 16; ++i) {
      int kk = c * 32 + ((i < 8) ? i : i + 8) + 8 * hi;
      _Float16 av = (_Float16)0.f;
      if (kk < KTOT) {                  // constant-foldable per (c,i)
        int ci  = kk % Cin;             // constant divisor -> mul/shift
        int t   = kk / Cin;
        int kw_ = t & (KW - 1);
        int kh_ = t >> 2;
        int ih  = oh * S + kh_ - PAD;
        int iw  = ow * S + kw_ - PAD;
        bool valid = ((unsigned)ih < (unsigned)Hin) && ((unsigned)iw < (unsigned)Win);
        const float* src =
            valid ? (x + ((size_t)(bi * Hin + ih) * Win + iw) * Cin + ci) : zpadf;
        av = (_Float16)(*src);          // unconditional load (pad block = zeros)
      }
      a[i] = av;
    }
    v16h b = wrow[(size_t)c * 32];
    if (c == 0)
      acc0 = __builtin_amdgcn_wmma_f32_16x16x32_f16(false, a, false, b, (short)0, acc0, false, false);
    else
      acc1 = __builtin_amdgcn_wmma_f32_16x16x32_f16(false, a, false, b, (short)0, acc1, false, false);
  }

  float bb = bias[n];
#pragma unroll
  for (int r = 0; r < 8; ++r) {
    int   m = tileM * 16 + r + (hi << 3);
    float v = fmaxf(acc0[r] + acc1[r] + bb, 0.f);   // relu
    outh[(size_t)m * Cout + n] = (_Float16)v;
  }
}

// zero the 256B OOB pad block (both f16 and f32 views read zeros)
__global__ void zpad_init_kernel(float* zp) {
  zp[threadIdx.x] = 0.f;   // 64 threads x 4B = 256B
}

// ---------------------------------------------------------------------------
// Repack f32 weights [KH,KW,Cin,Cout] -> fragment-ordered f16, linear-K chunks.
// wp[tileN][chunk][lane] = v16h; lane l: n = tileN*16 + (l&15);
// element i -> kk = chunk*32 + 8*(l>>4) + (i<8 ? i : i+8); kk -> (kh,kw,ci) row-major.
// (For Cin%32==0 this equals tap-major chunking used by conv_fast_kernel.)
// ---------------------------------------------------------------------------
__global__ void repack_w_kernel(const float* __restrict__ w, v16h* __restrict__ wp,
                                int Cin, int Cout, int KH, int KW, int ntiles)
{
  const int KTOT   = KH * KW * Cin;
  const int chunks = (KTOT + 31) >> 5;
  const int total  = ntiles * chunks * 32;
  int id = blockIdx.x * blockDim.x + threadIdx.x;
  if (id >= total) return;
  int l  = id & 31;
  int c  = (id >> 5) % chunks;
  int t  = id / (chunks * 32);
  int hi = l >> 4;
  int nn = t * 16 + (l & 15);
  int k0 = c * 32 + 8 * hi;
  v16h v;
#pragma unroll
  for (int i = 0; i < 16; ++i) {
    int   kk = k0 + ((i < 8) ? i : i + 8);
    float f  = 0.f;
    if (kk < KTOT && nn < Cout) {
      int ci  = kk % Cin;
      int tt  = kk / Cin;
      int kw_ = tt % KW;
      int kh_ = tt / KW;
      f = w[((kh_ * KW + kw_) * Cin + ci) * Cout + nn];
    }
    v[i] = (_Float16)f;
  }
  wp[id] = v;
}

// codebook: e16 = -2*emb (f16), e2[k] = |emb_k|^2
__global__ void emb_pack_kernel(const float* __restrict__ emb,
                                _Float16* __restrict__ e16, float* __restrict__ e2,
                                int K, int D)
{
  int k = blockIdx.x * blockDim.x + threadIdx.x;
  if (k < K) {
    float s = 0.f;
    for (int d = 0; d < D; ++d) {
      float v = emb[k * D + d];
      s += v * v;
      e16[k * D + d] = (_Float16)(-2.f * v);
    }
    e2[k] = s;
  }
}

// ---------------------------------------------------------------------------
// VQ argmin: scores = -2 z@emb^T + |e|^2 via WMMA on f16 operands.
// Two INDEPENDENT accumulators per code tile (summed in epilogue) -> no
// WMMA->WMMA RAW hazard. A frags loaded once (4 b128); B frags contiguous.
// ---------------------------------------------------------------------------
__global__ void vq_argmin_kernel(const _Float16* __restrict__ zh,
                                 const _Float16* __restrict__ e16,
                                 const float* __restrict__ e2,
                                 int* __restrict__ idx)
{
  const int lane = threadIdx.x & 31;
  const int wid  = threadIdx.x >> 5;
  const int hi   = lane >> 4;
  const int mr   = lane & 15;
  const int tileM = blockIdx.x * WAVES + wid;
  const int m     = tileM * 16 + mr;

  const _Float16* zr = zh + (size_t)m * 64;
  v8h l0 = *(const v8h*)(zr + 16 * hi);
  v8h h0 = *(const v8h*)(zr + 8 + 16 * hi);
  v8h l1 = *(const v8h*)(zr + 32 + 16 * hi);
  v8h h1 = *(const v8h*)(zr + 40 + 16 * hi);
  v16h a0, a1;
#pragma unroll
  for (int i = 0; i < 8; ++i) {
    a0[i] = l0[i]; a0[i + 8] = h0[i];
    a1[i] = l1[i]; a1[i + 8] = h1[i];
  }

  float best[8];
  int   bidx[8];
#pragma unroll
  for (int r = 0; r < 8; ++r) { best[r] = 3.4e38f; bidx[r] = 0; }

  for (int t = 0; t < 32; ++t) {                 // 32 tiles x 16 codes = 512
    int code = t * 16 + mr;
    const _Float16* er = e16 + (size_t)code * 64;
    v16h b0 = *(const v16h*)(er + 16 * hi);      // K = 16*hi + i (d 0..31)
    v16h b1 = *(const v16h*)(er + 32 + 16 * hi); // d 32..63
    v8f p0 = {}, p1 = {};
    p0 = __builtin_amdgcn_wmma_f32_16x16x32_f16(false, a0, false, b0, (short)0, p0, false, false);
    p1 = __builtin_amdgcn_wmma_f32_16x16x32_f16(false, a1, false, b1, (short)0, p1, false, false);

    float e2n = e2[code];
#pragma unroll
    for (int r = 0; r < 8; ++r) {
      float v = p0[r] + p1[r] + e2n;
      if (v < best[r]) { best[r] = v; bidx[r] = code; }
    }
  }

#pragma unroll
  for (int off = 8; off > 0; off >>= 1) {
#pragma unroll
    for (int r = 0; r < 8; ++r) {
      float ov = __shfl_xor(best[r], off, 32);
      int   oi = __shfl_xor(bidx[r], off, 32);
      if (ov < best[r] || (ov == best[r] && oi < bidx[r])) { best[r] = ov; bidx[r] = oi; }
    }
  }
  if (mr == 0) {
#pragma unroll
    for (int r = 0; r < 8; ++r)
      idx[tileM * 16 + r + (hi << 3)] = bidx[r];
  }
}

// q = emb[idx]: f32 (for loss) + f16 (decoder input) + idx copy-out
__global__ void vq_gather_kernel(const int* __restrict__ idx,
                                 const float* __restrict__ emb,
                                 float* __restrict__ qf, _Float16* __restrict__ qh,
                                 int* __restrict__ idx_out, int M, int D)
{
  int i = blockIdx.x * blockDim.x + threadIdx.x;
  if (i < M * D) {
    int   m = i / D, d = i - (i / D) * D;
    float v = emb[idx[m] * D + d];
    qf[i] = v;
    qh[i] = (_Float16)v;
  }
  if (i < M) idx_out[i] = idx[i];
}

__global__ void loss_init_kernel(float* loss) { *loss = 0.f; }

__global__ void vq_loss_kernel(const float* __restrict__ z, const float* __restrict__ q,
                               float* __restrict__ loss, int n, float scale)
{
  __shared__ float sdata[256];
  int   tid = threadIdx.x;
  float s   = 0.f;
  for (int i = blockIdx.x * blockDim.x + tid; i < n; i += gridDim.x * blockDim.x) {
    float d = q[i] - z[i];
    s += d * d;
  }
  sdata[tid] = s;
  __syncthreads();
  for (int o = 128; o > 0; o >>= 1) {
    if (tid < o) sdata[tid] += sdata[tid + o];
    __syncthreads();
  }
  if (tid == 0) atomicAdd(loss, sdata[0] * scale);
}

// ---------------------------------------------------------------------------
static void launch_fast(const _Float16* xh, const v16h* wp, const float* b,
                        const _Float16* zpad, _Float16* outh, float* outf,
                        int Hin, int Win, int Cin, int Hout, int Wout, int Cout,
                        int KH, int KW, int S, int pad, int transpose, int act,
                        hipStream_t st)
{
  int M      = 128 * Hout * Wout;
  int ntiles = (Cout + 15) / 16;
  int ntl    = ntiles < 4 ? ntiles : 4;         // all layers: ntiles in {1,2,4,8}
  dim3 grid(M / 16 / WAVES, ntiles / ntl);
  dim3 blk(WAVES * 32);
  switch (ntl) {
    case 1:
      conv_fast_kernel<1><<<grid, blk, 0, st>>>(xh, wp, b, zpad, outh, outf,
          Hin, Win, Cin, Hout, Wout, Cout, KH, KW, S, pad, transpose, act);
      break;
    case 2:
      conv_fast_kernel<2><<<grid, blk, 0, st>>>(xh, wp, b, zpad, outh, outf,
          Hin, Win, Cin, Hout, Wout, Cout, KH, KW, S, pad, transpose, act);
      break;
    default:
      conv_fast_kernel<4><<<grid, blk, 0, st>>>(xh, wp, b, zpad, outh, outf,
          Hin, Win, Cin, Hout, Wout, Cout, KH, KW, S, pad, transpose, act);
      break;
  }
}

static void launch_repack(const float* w, v16h* wp, int Cin, int Cout, int KH, int KW,
                          hipStream_t st)
{
  int ntiles = (Cout + 15) / 16;
  int chunks = (KH * KW * Cin + 31) / 32;
  int total  = ntiles * chunks * 32;
  repack_w_kernel<<<(total + 255) / 256, 256, 0, st>>>(w, wp, Cin, Cout, KH, KW, ntiles);
}

extern "C" void kernel_launch(void* const* d_in, const int* in_sizes, int n_in,
                              void* d_out, int out_size, void* d_ws, size_t ws_size,
                              hipStream_t stream)
{
  (void)in_sizes; (void)n_in; (void)out_size; (void)ws_size;

  const float* x      = (const float*)d_in[0];
  const float* enc_w1 = (const float*)d_in[1];  const float* enc_b1 = (const float*)d_in[2];
  const float* enc_w2 = (const float*)d_in[3];  const float* enc_b2 = (const float*)d_in[4];
  const float* enc_w3 = (const float*)d_in[5];  const float* enc_b3 = (const float*)d_in[6];
  const float* enc_w4 = (const float*)d_in[7];  const float* enc_b4 = (const float*)d_in[8];
  const float* emb    = (const float*)d_in[9];
  const float* dec_w1 = (const float*)d_in[10]; const float* dec_b1 = (const float*)d_in[11];
  const float* dec_w2 = (const float*)d_in[12]; const float* dec_b2 = (const float*)d_in[13];
  const float* dec_w3 = (const float*)d_in[14]; const float* dec_b3 = (const float*)d_in[15];
  const float* dec_w4 = (const float*)d_in[16]; const float* dec_b4 = (const float*)d_in[17];

  // ---- byte-based workspace carve-up (256B aligned blocks) ----
  char* base = (char*)d_ws;
  auto alloc = [&](size_t bytes) -> char* {
    char* p = base;
    base += (bytes + 255) & ~(size_t)255;
    return p;
  };
  _Float16* a1h  = (_Float16*)alloc((size_t)16777216 * 2);  // [128,64,64,32]
  _Float16* a2h  = (_Float16*)alloc((size_t)8388608 * 2);   // [128,32,32,64]
  _Float16* a3h  = (_Float16*)alloc((size_t)4194304 * 2);   // [128,16,16,128]
  _Float16* zh   = (_Float16*)alloc((size_t)2097152 * 2);   // [128,16,16,64]
  float*    zf   = (float*)   alloc((size_t)2097152 * 4);
  _Float16* qh   = (_Float16*)alloc((size_t)2097152 * 2);
  float*    qf   = (float*)   alloc((size_t)2097152 * 4);
  _Float16* e16  = (_Float16*)alloc((size_t)512 * 64 * 2);
  float*    e2   = (float*)   alloc((size_t)512 * 4);
  int*      vidx = (int*)     alloc((size_t)32768 * 4);
  float*    zpad = (float*)   alloc((size_t)256);           // zeroed OOB pad block
  v16h* wp1  = (v16h*)alloc((size_t)128 * 32);   // conv1: 2 tiles * 2 chunks * 32
  v16h* wp2  = (v16h*)alloc((size_t)2048 * 32);  // conv2: 4 * 16 * 32
  v16h* wp3  = (v16h*)alloc((size_t)8192 * 32);  // conv3: 8 * 32 * 32
  v16h* wp4  = (v16h*)alloc((size_t)2048 * 32);  // conv4: 4 * 16 * 32
  v16h* wpd1 = (v16h*)alloc((size_t)2048 * 32);  // dec1:  8 *  8 * 32
  v16h* wpd2 = (v16h*)alloc((size_t)8192 * 32);  // dec2:  4 * 64 * 32
  v16h* wpd3 = (v16h*)alloc((size_t)2048 * 32);  // dec3:  2 * 32 * 32
  v16h* wpd4 = (v16h*)alloc((size_t)512 * 32);   // dec4:  1 * 16 * 32

  float* out_f    = (float*)d_out;               // y: 6291456 f32
  float* loss_ptr = out_f + 6291456;
  int*   idx_out  = (int*)out_f + 6291456 + 1;
  const _Float16* zpadh = (const _Float16*)zpad;

  // ---- init + weight / codebook repacks (depend only on inputs) ----
  zpad_init_kernel<<<1, 64, 0, stream>>>(zpad);
  launch_repack(enc_w1, wp1,   3,  32, 4, 4, stream);
  launch_repack(enc_w2, wp2,  32,  64, 4, 4, stream);
  launch_repack(enc_w3, wp3,  64, 128, 4, 4, stream);
  launch_repack(enc_w4, wp4, 128,  64, 2, 2, stream);
  launch_repack(dec_w1, wpd1, 64, 128, 2, 2, stream);
  launch_repack(dec_w2, wpd2,128,  64, 4, 4, stream);
  launch_repack(dec_w3, wpd3, 64,  32, 4, 4, stream);
  launch_repack(dec_w4, wpd4, 32,   3, 4, 4, stream);
  emb_pack_kernel<<<2, 256, 0, stream>>>(emb, e16, e2, 512, 64);

  // ---- encoder ----
  {
    dim3 grid(128 * 64 * 64 / 16 / WAVES, 2);
    conv1_kernel<<<grid, WAVES * 32, 0, stream>>>(x, wp1, enc_b1, zpad, a1h);
  }
  launch_fast(a1h, wp2, enc_b2, zpadh, a2h, nullptr, 64, 64, 32, 32, 32,  64, 4, 4, 2, 1, 0, 1, stream);
  launch_fast(a2h, wp3, enc_b3, zpadh, a3h, nullptr, 32, 32, 64, 16, 16, 128, 4, 4, 2, 1, 0, 1, stream);
  launch_fast(a3h, wp4, enc_b4, zpadh, zh,  zf,      16, 16,128, 16, 16,  64, 2, 2, 1, 0, 0, 0, stream);

  // ---- vector quantizer ----
  vq_argmin_kernel<<<2048 / WAVES, WAVES * 32, 0, stream>>>(zh, e16, e2, vidx);
  loss_init_kernel<<<1, 1, 0, stream>>>(loss_ptr);
  vq_gather_kernel<<<(2097152 + 255) / 256, 256, 0, stream>>>(vidx, emb, qf, qh, idx_out, 32768, 64);
  vq_loss_kernel<<<512, 256, 0, stream>>>(zf, qf, loss_ptr, 2097152, 0.25f / 2097152.0f);

  // ---- decoder (transposed convs) ----
  launch_fast(qh,  wpd1, dec_b1, zpadh, a3h, nullptr, 16, 16,  64, 16, 16, 128, 2, 2, 1, 1, 1, 1, stream);
  launch_fast(a3h, wpd2, dec_b2, zpadh, a2h, nullptr, 16, 16, 128, 32, 32,  64, 4, 4, 2, 1, 1, 1, stream);
  launch_fast(a2h, wpd3, dec_b3, zpadh, a1h, nullptr, 32, 32,  64, 64, 64,  32, 4, 4, 2, 1, 1, 1, stream);
  launch_fast(a1h, wpd4, dec_b4, zpadh, nullptr, out_f, 64, 64, 32, 128, 128, 3, 4, 4, 2, 1, 1, 2, stream);
}
